// Head_48172353192210
// MI455X (gfx1250) — matrix operand order repaired and checked by reading, not compiled, not run on hardware
//
#include <hip/hip_runtime.h>
#include <stdint.h>

#define B_ 8
#define T_ 2048
#define D_ 384
#define K_ 64

typedef __attribute__((ext_vector_type(16))) _Float16 v16h;
typedef __attribute__((ext_vector_type(8)))  _Float16 v8h;
typedef __attribute__((ext_vector_type(4)))  _Float16 v4h;
typedef __attribute__((ext_vector_type(8)))  float    v8f;
typedef __attribute__((ext_vector_type(4)))  float    v4f;
typedef __attribute__((ext_vector_type(4)))  int      i4v;

// ---- gfx1250 async global->LDS copy path (compile-safe guard) ----
#if defined(__has_builtin)
#if __has_builtin(__builtin_amdgcn_global_load_async_to_lds_b128)
#define HAVE_ASYNC_LDS 1
#endif
#if __has_builtin(__builtin_amdgcn_tensor_load_to_lds)
#define HAVE_TDM 1
#endif
#endif

#ifdef HAVE_ASYNC_LDS
// builtin: (int4* src /*generic*/, int4 AS(3)* dst, imm offset, imm cpol)
#define GBL_AS(p) ((i4v*)(uintptr_t)(p))
#define LDS_AS(p) ((__attribute__((address_space(3))) i4v*)(uint32_t)(uintptr_t)(p))
static __device__ __forceinline__ void wait_async0() {
#if __has_builtin(__builtin_amdgcn_s_wait_asynccnt)
  __builtin_amdgcn_s_wait_asynccnt(0);
#else
  asm volatile("s_wait_asynccnt 0x0" ::: "memory");
#endif
}
#endif

#ifdef HAVE_TDM
typedef __attribute__((ext_vector_type(4))) unsigned int u32x4;
typedef __attribute__((ext_vector_type(8))) int          i32x8;
typedef __attribute__((ext_vector_type(4))) int          i32x4;

// Build a 2D-tile Tensor-DMA descriptor (ISA 8.3/8.4) and issue
// TENSOR_LOAD_TO_LDS.  f16 elements (data_size=2B).  LDS row padding:
// pad_interval_code=4 -> pad every 32 DWORDs (128B row),
// pad_amount_code=3 -> insert 4 DWORDs (8 f16) => LDS row stride 72 f16.
static __device__ __forceinline__ void tdm_load_2d_f16(
    const _Float16* gsrc, const _Float16* ldst,
    unsigned tensor_d0, unsigned tensor_d1,
    unsigned tile_d0, unsigned tile_d1,
    unsigned d0_stride, int pad_enable,
    unsigned pad_interval_code, unsigned pad_amount_code) {
  uint64_t ga = (uint64_t)(uintptr_t)gsrc;
  u32x4 g0;
  g0[0] = 1u;                                                // count=1 (valid)
  g0[1] = (uint32_t)(uintptr_t)ldst;                         // lds_addr (bytes)
  g0[2] = (uint32_t)ga;                                      // global_addr[31:0]
  g0[3] = (uint32_t)((ga >> 32) & 0x01FFFFFFu) | (2u << 30); // ga[56:32] | type=2
  i32x8 g1;
  g1[0] = (int)((1u << 16) |                                 // data_size = 2B
                (pad_enable ? (1u << 20) : 0u) |
                (pad_enable ? (pad_interval_code << 22) : 0u) |
                (pad_enable ? (pad_amount_code << 25) : 0u));
  g1[1] = (int)((tensor_d0 & 0xFFFFu) << 16);                // tensor_dim0 lo
  g1[2] = (int)((tensor_d0 >> 16) | ((tensor_d1 & 0xFFFFu) << 16));
  g1[3] = (int)((tensor_d1 >> 16) | ((tile_d0 & 0xFFFFu) << 16));
  g1[4] = (int)(tile_d1 & 0xFFFFu);                          // tile_dim2 = 0
  g1[5] = (int)d0_stride;                                    // dim0_stride lo
  g1[6] = 0;                                                 // stride hi / dim1_stride
  g1[7] = 0;
  i32x4 gz4 = {};
  i32x8 gz8 = {};
  // 6-arg toolchain form: (g0, g1, g2, g3, extra_group, cpol)
  __builtin_amdgcn_tensor_load_to_lds(g0, g1, gz4, gz4, gz8, 0);
}
#endif

// ---- WMMA fragment builders (wave32, contiguous k = 16*half + s in both A and B) ----
// All LDS operand tiles are stored K-dim-contiguous so fragments load as b128.

static __device__ __forceinline__ v16h cat8(v8h lo, v8h hi) {
  return __builtin_shufflevector(lo, hi, 0, 1, 2, 3, 4, 5, 6, 7,
                                 8, 9, 10, 11, 12, 13, 14, 15);
}

// A matrix 16x32 f16 from [M][Kdim] storage (stride multiple of 8 f16).
static __device__ __forceinline__ v16h load_A_f16(const _Float16* base, int stride,
                                                  int m0, int k0, int lane) {
  const int m = m0 + (lane & 15);
  const int half = lane >> 4;
  const _Float16* p = base + m * stride + k0 + (half << 4);
  v8h lo = *(const v8h*)(p);
  v8h hi = *(const v8h*)(p + 8);
  return cat8(lo, hi);
}

// B matrix 32x16 f16 with operand stored as [N][Kdim] (K contiguous).
static __device__ __forceinline__ v16h load_B_nk(const _Float16* base, int stride,
                                                 int k0, int n0, int lane) {
  const int n = n0 + (lane & 15);
  const int half = lane >> 4;
  const _Float16* p = base + n * stride + k0 + (half << 4);
  v8h lo = *(const v8h*)(p);
  v8h hi = *(const v8h*)(p + 8);
  return cat8(lo, hi);
}

// ================= Kernel 1: fused QKV projection =================
// cols 0..63 -> q [row][k], 64..127 -> k [row][k], 128..511 -> v stored
// TRANSPOSED as vt[b][d][t] so the attention kernel can TDM 2D tiles.
__global__ __launch_bounds__(256) void qkv_proj_kernel(
    const float* __restrict__ x, const float* __restrict__ Wq,
    const float* __restrict__ Wk, const float* __restrict__ Wv,
    _Float16* __restrict__ qws, _Float16* __restrict__ kws,
    _Float16* __restrict__ vtws) {
  __shared__ _Float16 Xs[128 * 40];   // 128x32 tile [m][k], stride 40
  __shared__ _Float16 Wt[64 * 40];    // 64x32 tile  [n][k], transposed, stride 40

  const int tid  = threadIdx.x;
  const int lane = tid & 31;
  const int wave = tid >> 5;           // 8 waves; wave = M subtile
  const int mblock = blockIdx.x;       // 128 rows of flattened B*T
  const int nblock = blockIdx.y;       // 64-wide column block of 512

  // block-uniform weight-region select (keeps staging branch-free)
  const float* wsrc; int wld, coff;
  if (nblock == 0)      { wsrc = Wq; wld = K_; coff = 0; }
  else if (nblock == 1) { wsrc = Wk; wld = K_; coff = 0; }
  else                  { wsrc = Wv; wld = D_; coff = (nblock - 2) * 64; }

  v8f z = {};
  v8f acc[4];
#pragma unroll
  for (int i = 0; i < 4; ++i) acc[i] = z;

  for (int k0 = 0; k0 < D_; k0 += 32) {
    __syncthreads();
    // X tile: 1024 float4 loads -> f16 -> b64 LDS stores
#pragma unroll
    for (int i = 0; i < 4; ++i) {
      int idx = tid + i * 256;
      int r = idx >> 3, c4 = (idx & 7) * 4;
      v4f xv = *(const v4f*)(x + (size_t)(mblock * 128 + r) * D_ + k0 + c4);
      v4h hv;
#pragma unroll
      for (int j = 0; j < 4; ++j) hv[j] = (_Float16)xv[j];
      *(v4h*)(&Xs[r * 40 + c4]) = hv;
    }
    // W tile 32x64: float4 loads, transposed scalar stores [n][d]
#pragma unroll
    for (int i = 0; i < 2; ++i) {
      int idx = tid + i * 256;              // 512 float4
      int d = idx >> 4, n4 = (idx & 15) * 4;
      v4f wv = *(const v4f*)(wsrc + (size_t)(k0 + d) * wld + coff + n4);
#pragma unroll
      for (int j = 0; j < 4; ++j) Wt[(n4 + j) * 40 + d] = (_Float16)wv[j];
    }
    __syncthreads();

    v16h a = load_A_f16(Xs, 40, wave * 16, 0, lane);
#pragma unroll
    for (int nt = 0; nt < 4; ++nt) {
      v16h b = load_B_nk(Wt, 40, 0, nt * 16, lane);
      acc[nt] = __builtin_amdgcn_wmma_f32_16x16x32_f16(
          false, a, false, b, (short)0, acc[nt], false, false);
    }
  }

  const int half = lane >> 4;
  if (nblock < 2) {
    _Float16* dst = (nblock == 0) ? qws : kws;
#pragma unroll
    for (int nt = 0; nt < 4; ++nt) {
#pragma unroll
      for (int r = 0; r < 8; ++r) {
        int row = mblock * 128 + wave * 16 + r + half * 8;
        int cin = nt * 16 + (lane & 15);
        dst[(size_t)row * K_ + cin] = (_Float16)acc[nt][r];
      }
    }
  } else {
    // transposed scatter: vt[b][d][t]  (128-row block lies within one batch)
#pragma unroll
    for (int nt = 0; nt < 4; ++nt) {
#pragma unroll
      for (int r = 0; r < 8; ++r) {
        int row = mblock * 128 + wave * 16 + r + half * 8;
        int d = coff + nt * 16 + (lane & 15);
        int bb = row >> 11, tt = row & (T_ - 1);
        vtws[((size_t)bb * D_ + d) * T_ + tt] = (_Float16)acc[nt][r];
      }
    }
  }
}

// ================= Kernel 2: flash attention (causal) =================
// Dynamic LDS layout, DOUBLE-BUFFERED K/V tiles (165632 B < 320KB/WGP):
#define SMEM_SS    0         // 64*68 f32  = 17408 B  (scores, padded)
#define SMEM_STATS 17408     // 3*64 f32   = 768  B   (m, l, alpha)
#define SMEM_QS    18176     // 64*72 f16  = 9216 B   [query][k]
#define SMEM_PS    27392     // 64*72 f16  = 9216 B   [query][key]
#define SMEM_KS0   36608     // 64*72 f16  = 9216 B   [key][k]     buffer 0
#define KS_BYTES   9216
#define SMEM_VT0   55040     // 384*72 f16 = 55296 B  [d][key]     buffer 0
#define VT_BYTES   55296
#define SMEM_TOTAL 165632

__global__ __launch_bounds__(256) void flash_attn_kernel(
    const _Float16* __restrict__ qws, const _Float16* __restrict__ kws,
    const _Float16* __restrict__ vtws, float* __restrict__ out) {
  extern __shared__ char smem[];
  float*    Ss   = (float*)(smem + SMEM_SS);
  float*    mrow = (float*)(smem + SMEM_STATS);
  float*    lrow = mrow + 64;
  float*    arow = mrow + 128;
  _Float16* Qs = (_Float16*)(smem + SMEM_QS);
  _Float16* Ps = (_Float16*)(smem + SMEM_PS);

  const int tid  = threadIdx.x;
  const int lane = tid & 31;
  const int wave = tid >> 5;
  const int half = lane >> 4;
  const int qb = blockIdx.x;     // 64-query block
  const int b  = blockIdx.y;     // batch

  const int mt = wave & 3;       // this wave's O row-tile
  const int nh = wave >> 2;      // O column half: d-tiles [0..11] or [12..23]

  v8f z = {};
  v8f accO[12];
#pragma unroll
  for (int i = 0; i < 12; ++i) accO[i] = z;

  // ---- prologue: Q block via async b128; first K/V tiles via TDM ----
#pragma unroll
  for (int i = 0; i < 2; ++i) {
    int idx = tid + i * 256;
    int r = idx >> 3, c8 = (idx & 7) * 8;
    const _Float16* src = qws + (size_t)(b * T_ + qb * 64 + r) * K_ + c8;
    _Float16* dst = &Qs[r * 72 + c8];
#ifdef HAVE_ASYNC_LDS
    __builtin_amdgcn_global_load_async_to_lds_b128(GBL_AS(src), LDS_AS(dst), 0, 0);
#else
    *(v8h*)dst = *(const v8h*)src;
#endif
  }
  {
    _Float16* Ks0 = (_Float16*)(smem + SMEM_KS0);
    _Float16* Vt0 = (_Float16*)(smem + SMEM_VT0);
#ifdef HAVE_TDM
    if (wave == 0) {
      tdm_load_2d_f16(kws + (size_t)(b * T_) * K_, Ks0,
                      64, 64, 64, 64, 64, 1, 4, 3);
      tdm_load_2d_f16(vtws + (size_t)b * D_ * T_, Vt0,
                      T_, D_, 64, D_, T_, 1, 4, 3);
    }
#else
#pragma unroll
    for (int i = 0; i < 2; ++i) {
      int idx = tid + i * 256;
      int r = idx >> 3, c8 = (idx & 7) * 8;
      *(v8h*)(&Ks0[r * 72 + c8]) =
          *(const v8h*)(kws + (size_t)(b * T_ + r) * K_ + c8);
    }
#pragma unroll
    for (int i = 0; i < 12; ++i) {
      int idx = tid + i * 256;
      int d = idx >> 3, c8 = (idx & 7) * 8;
      *(v8h*)(&Vt0[d * 72 + c8]) =
          *(const v8h*)(vtws + ((size_t)b * D_ + d) * T_ + c8);
    }
#endif
  }
  if (tid < 64) { mrow[tid] = -1e30f; lrow[tid] = 0.f; }
#ifdef HAVE_ASYNC_LDS
  wait_async0();
#endif
#ifdef HAVE_TDM
  if (wave == 0) __builtin_amdgcn_s_wait_tensorcnt(0);
#endif
  __syncthreads();

  for (int sb = 0; sb <= qb; ++sb) {
    const int cur = sb & 1;
    const _Float16* Ks = (const _Float16*)(smem + SMEM_KS0 + cur * KS_BYTES);
    const _Float16* Vt = (const _Float16*)(smem + SMEM_VT0 + cur * VT_BYTES);
    _Float16* Ksn = (_Float16*)(smem + SMEM_KS0 + (cur ^ 1) * KS_BYTES);
    _Float16* Vtn = (_Float16*)(smem + SMEM_VT0 + (cur ^ 1) * VT_BYTES);

    // ---- kick off DMA of the NEXT K/V block into the other buffer ----
    if (sb < qb) {
#ifdef HAVE_TDM
      if (wave == 0) {
        tdm_load_2d_f16(kws + (size_t)(b * T_ + (sb + 1) * 64) * K_, Ksn,
                        64, 64, 64, 64, 64, 1, 4, 3);
        tdm_load_2d_f16(vtws + (size_t)b * D_ * T_ + (size_t)(sb + 1) * 64,
                        Vtn, T_, D_, 64, D_, T_, 1, 4, 3);
      }
#else
#pragma unroll
      for (int i = 0; i < 2; ++i) {
        int idx = tid + i * 256;
        int r = idx >> 3, c8 = (idx & 7) * 8;
        *(v8h*)(&Ksn[r * 72 + c8]) =
            *(const v8h*)(kws + (size_t)(b * T_ + (sb + 1) * 64 + r) * K_ + c8);
      }
#pragma unroll
      for (int i = 0; i < 12; ++i) {
        int idx = tid + i * 256;
        int d = idx >> 3, c8 = (idx & 7) * 8;
        *(v8h*)(&Vtn[d * 72 + c8]) =
            *(const v8h*)(vtws + ((size_t)b * D_ + d) * T_ + (sb + 1) * 64 + c8);
      }
#endif
      // warm L2 one block ahead of the DMA
      if (sb + 2 <= qb) {
        __builtin_prefetch(kws + (size_t)(b * T_ + (sb + 2) * 64 + (tid & 63)) * K_, 0, 0);
        __builtin_prefetch(vtws + ((size_t)b * D_ + tid) * T_ + (sb + 2) * 64, 0, 0);
        if (tid < D_ - 256)
          __builtin_prefetch(vtws + ((size_t)b * D_ + 256 + tid) * T_ + (sb + 2) * 64, 0, 0);
      }
    }

    // ---- S = Q*K^T : 4x4 16x16 tiles, 2 per wave; K-dim 64 = 2 WMMA steps ----
#pragma unroll
    for (int ti = 0; ti < 2; ++ti) {
      int t = wave * 2 + ti;
      int mts = t >> 2, nts = t & 3;
      v8f s = z;
#pragma unroll
      for (int kc = 0; kc < 2; ++kc) {
        v16h a  = load_A_f16(Qs, 72, mts * 16, kc * 32, lane);
        v16h bt = load_B_nk(Ks, 72, kc * 32, nts * 16, lane);
        s = __builtin_amdgcn_wmma_f32_16x16x32_f16(
            false, a, false, bt, (short)0, s, false, false);
      }
#pragma unroll
      for (int r = 0; r < 8; ++r) {
        int lr = mts * 16 + r + half * 8;
        int lc = nts * 16 + (lane & 15);
        int rg = qb * 64 + lr, cg = sb * 64 + lc;
        Ss[lr * 68 + lc] = (cg <= rg) ? s[r] * 0.125f : -1e30f;  // 1/sqrt(64)
      }
    }
    __syncthreads();

    // ---- online softmax: 4 lanes per row, shfl_xor reductions ----
    {
      int row = tid >> 2, quad = tid & 3;
      v4f sv[4];
#pragma unroll
      for (int q4 = 0; q4 < 4; ++q4)
        sv[q4] = *(const v4f*)(&Ss[row * 68 + quad * 16 + q4 * 4]);
      float mx = -1e30f;
#pragma unroll
      for (int j = 0; j < 16; ++j) mx = fmaxf(mx, sv[j >> 2][j & 3]);
      mx = fmaxf(mx, __shfl_xor(mx, 1, 32));
      mx = fmaxf(mx, __shfl_xor(mx, 2, 32));
      float mold = mrow[row];
      float mnew = fmaxf(mold, mx);
      float alpha = __expf(mold - mnew);
      float sum = 0.f;
      v8h p0, p1;
#pragma unroll
      for (int j = 0; j < 8; ++j) {
        float p = __expf(sv[j >> 2][j & 3] - mnew);
        p0[j] = (_Float16)p; sum += p;
      }
#pragma unroll
      for (int j = 0; j < 8; ++j) {
        float p = __expf(sv[(j + 8) >> 2][(j + 8) & 3] - mnew);
        p1[j] = (_Float16)p; sum += p;
      }
      *(v8h*)(&Ps[row * 72 + quad * 16])     = p0;
      *(v8h*)(&Ps[row * 72 + quad * 16 + 8]) = p1;
      sum += __shfl_xor(sum, 1, 32);
      sum += __shfl_xor(sum, 2, 32);
      if (quad == 0) {
        mrow[row] = mnew;
        lrow[row] = lrow[row] * alpha + sum;
        arow[row] = alpha;
      }
    }
    __syncthreads();

    // ---- O = alpha*O + P*V over this wave's 12 d-tiles ----
    float al[8];
#pragma unroll
    for (int r = 0; r < 8; ++r) al[r] = arow[mt * 16 + r + half * 8];
    v16h pa0 = load_A_f16(Ps, 72, mt * 16, 0, lane);
    v16h pa1 = load_A_f16(Ps, 72, mt * 16, 32, lane);
#pragma unroll
    for (int nt = 0; nt < 12; ++nt) {
#pragma unroll
      for (int r = 0; r < 8; ++r) accO[nt][r] *= al[r];
      int n0 = (nh * 12 + nt) * 16;
      v16h bv0 = load_B_nk(Vt, 72, 0, n0, lane);
      accO[nt] = __builtin_amdgcn_wmma_f32_16x16x32_f16(
          false, pa0, false, bv0, (short)0, accO[nt], false, false);
      v16h bv1 = load_B_nk(Vt, 72, 32, n0, lane);
      accO[nt] = __builtin_amdgcn_wmma_f32_16x16x32_f16(
          false, pa1, false, bv1, (short)0, accO[nt], false, false);
    }

    // ---- drain the in-flight DMA before flipping buffers ----
#ifdef HAVE_TDM
    if (wave == 0 && sb < qb) __builtin_amdgcn_s_wait_tensorcnt(0);
#endif
    __syncthreads();
  }

  // ---- epilogue: O /= l, store fp32 ----
  float linv[8];
#pragma unroll
  for (int r = 0; r < 8; ++r) linv[r] = 1.0f / lrow[mt * 16 + r + half * 8];
#pragma unroll
  for (int nt = 0; nt < 12; ++nt) {
    int col = (nh * 12 + nt) * 16 + (lane & 15);
#pragma unroll
    for (int r = 0; r < 8; ++r) {
      int row = qb * 64 + mt * 16 + r + half * 8;
      out[(size_t)(b * T_ + row) * D_ + col] = accO[nt][r] * linv[r];
    }
  }
}

extern "C" void kernel_launch(void* const* d_in, const int* in_sizes, int n_in,
                              void* d_out, int out_size, void* d_ws, size_t ws_size,
                              hipStream_t stream) {
  (void)in_sizes; (void)n_in; (void)out_size; (void)ws_size;
  const float* x  = (const float*)d_in[0];
  const float* Wq = (const float*)d_in[1];
  const float* Wk = (const float*)d_in[2];
  const float* Wv = (const float*)d_in[3];
  float* out = (float*)d_out;

  _Float16* qws  = (_Float16*)d_ws;                    // B*T*64 f16 [row][k]
  _Float16* kws  = qws + (size_t)B_ * T_ * K_;         // B*T*64 f16 [row][k]
  _Float16* vtws = kws + (size_t)B_ * T_ * K_;         // B*D*T f16 [b][d][t]

  dim3 blk(256);
  dim3 g1(128, 8);                // (B*T)/128 row blocks x 512/64 col blocks
  qkv_proj_kernel<<<g1, blk, 0, stream>>>(x, Wq, Wk, Wv, qws, kws, vtws);

  dim3 g2(T_ / 64, B_);           // 32 query blocks x 8 batches
  flash_attn_kernel<<<g2, blk, SMEM_TOTAL, stream>>>(qws, kws, vtws, out);
}
